// DeepSeekMoE_66554813219270
// MI455X (gfx1250) — matrix operand order repaired and checked by reading, not compiled, not run on hardware
//
#include <hip/hip_runtime.h>
#include <stdint.h>

// ---------------------------------------------------------------------------
// DeepSeek MoE forward for MI455X (gfx1250, wave32, WMMA bf16 + async-LDS DMA)
//   out  = sharedSwiGLU(x) + sum_e gatherSwiGLU_e(x) * combine_w
//   idx  = top-2 expert indices per token
//
// Pipeline per expert: [transpose+cvt weights -> bf16 [N][K]] ->
//   gemm1 (x@Wg, x@Wu fused, silu*up*combine -> bf16 h) -> gemm2 (h@Wd scatter)
// GEMM tiles are staged with GLOBAL_LOAD_ASYNC_TO_LDS_B128 (no VGPR traffic),
// LDS double-buffered, one barrier per K-tile, next tile's DMA overlaps WMMA.
// ---------------------------------------------------------------------------

typedef __attribute__((ext_vector_type(16))) __bf16 v16bf;
typedef __attribute__((ext_vector_type(8)))  float  v8f;

#define NEXP   8
#define BM     128
#define BK     32
#define BN1    64     // gemm1 N-tile (two weight matrices -> effectively 128)
#define BN2    128    // gemm2 N-tile
#define TPB    256
#define PADU   20     // LDS row pitch in u32: 32 bf16 (16 u32) + 4 u32 pad = 80B

// ---- CDNA5 async global->LDS staging (ASYNCcnt-tracked, reg-free) ----------
__device__ __forceinline__ void async_b128(void* lds, const void* g) {
  asm volatile("global_load_async_to_lds_b128 %0, %1, off"
               :: "v"((uint)(uintptr_t)lds), "v"(g)
               : "memory");
}
__device__ __forceinline__ void wait_async0() {
#if __has_builtin(__builtin_amdgcn_s_wait_asynccnt)
  __builtin_amdgcn_s_wait_asynccnt(0);
#else
  asm volatile("s_wait_asynccnt 0x0" ::: "memory");
#endif
}

// ---- LDS fragment loaders matching the CDNA5 16-bit WMMA VGPR layouts ------
// A (16x32 bf16): lanes 0-15 hold M=lane, K 0-7 then 16-23 (bytes 0 and 32);
//                 lanes 16-31 hold M=lane-16, K 8-15 then 24-31 (bytes 16, 48).
__device__ __forceinline__ v16bf frag_a(const uint* base, int row, int half) {
  union { v16bf v; uint4 q[2]; } f;
  const uint* p = base + row * PADU + half * 4;      // half*16 bytes
  f.q[0] = *(const uint4*)(p);
  f.q[1] = *(const uint4*)(p + 8);                   // +32 bytes
  return f.v;
}
// B (32x16 bf16), LDS stored [N][K]: lanes 0-15 hold N=lane, K 0-7 then 16-23;
// lanes 16-31 hold N=lane-16, K 8-15 then 24-31 (same K order as A).
__device__ __forceinline__ v16bf frag_b(const uint* base, int n, int half) {
  union { v16bf v; uint4 q[2]; } f;
  const uint* p = base + n * PADU + half * 4;        // half*16 bytes
  f.q[0] = *(const uint4*)(p);
  f.q[1] = *(const uint4*)(p + 8);
  return f.v;
}

__device__ __forceinline__ v8f wmma_bf16(v16bf a, v16bf b, v8f c) {
  return __builtin_amdgcn_wmma_f32_16x16x32_bf16(false, a, false, b,
                                                 (short)0, c, false, false);
}

__device__ __forceinline__ uint pack_bf16(float lo, float hi) {
#if __has_builtin(__builtin_amdgcn_cvt_pk_bf16_f32)
  typedef __attribute__((ext_vector_type(2))) __bf16 v2bf;
  union { v2bf v; uint u; } c;
  c.v = __builtin_amdgcn_cvt_pk_bf16_f32(lo, hi);
  return c.u;
#else
  union { __bf16 h2[2]; uint u; } pk;
  pk.h2[0] = (__bf16)lo; pk.h2[1] = (__bf16)hi;
  return pk.u;
#endif
}

// ---------------------------------------------------------------------------
// Router: scores = sigmoid(x @ gate_w^T); top-2 with bias + index tie-break;
// emits per-expert gather lists + normalized combine weights + index output.
// ---------------------------------------------------------------------------
__global__ __launch_bounds__(TPB)
void moe_router(const float* __restrict__ x, const float* __restrict__ gw,
                const float* __restrict__ bias, int* __restrict__ counts,
                int* __restrict__ list, float* __restrict__ wlist,
                int* __restrict__ idx_out, int T, int H)
{
  const int wv = threadIdx.x >> 5, lane = threadIdx.x & 31;
  const int t = blockIdx.x * (TPB / 32) + wv;
  if (t >= T) return;

  float acc[NEXP];
#pragma unroll
  for (int e = 0; e < NEXP; ++e) acc[e] = 0.f;

  for (int h = lane; h < H; h += 32) {
    const float xv = x[(size_t)t * H + h];
#pragma unroll
    for (int e = 0; e < NEXP; ++e)
      acc[e] += xv * gw[(size_t)e * H + h];
  }
#pragma unroll
  for (int e = 0; e < NEXP; ++e)
    for (int off = 16; off > 0; off >>= 1)
      acc[e] += __shfl_xor(acc[e], off, 32);

  if (lane == 0) {
    float sc[NEXP], sel[NEXP];
#pragma unroll
    for (int e = 0; e < NEXP; ++e) {
      sc[e]  = 1.f / (1.f + __expf(-acc[e]));
      sel[e] = sc[e] + bias[e] + (float)e * 1e-6f;
    }
    int i1 = 0;
#pragma unroll
    for (int e = 1; e < NEXP; ++e) if (sel[e] > sel[i1]) i1 = e;
    int i2 = (i1 == 0) ? 1 : 0;
#pragma unroll
    for (int e = 0; e < NEXP; ++e)
      if (e != i1 && sel[e] > sel[i2]) i2 = e;

    const float s1 = sc[i1], s2 = sc[i2];
    float denom = s1 + s2; if (denom < 1e-9f) denom = 1e-9f;
    const float w1 = s1 / denom, w2 = s2 / denom;

    idx_out[t * 2 + 0] = i1;
    idx_out[t * 2 + 1] = i2;

    int p1 = atomicAdd(&counts[i1], 1);
    list[(size_t)i1 * T + p1]  = t;
    wlist[(size_t)i1 * T + p1] = w1;
    int p2 = atomicAdd(&counts[i2], 1);
    list[(size_t)i2 * T + p2]  = t;
    wlist[(size_t)i2 * T + p2] = w2;
  }
}

__global__ void moe_zero_counts(int* counts) {
  if (threadIdx.x < NEXP) counts[threadIdx.x] = 0;
}

// ---------------------------------------------------------------------------
// fp32 [K][N] -> bf16 [N][K] transpose+convert (32x32 tiles via LDS).
// ---------------------------------------------------------------------------
__global__ __launch_bounds__(TPB)
void moe_cvt_tr(const float* __restrict__ in, unsigned short* __restrict__ outp,
                int K, int N)
{
  __shared__ float ts[32][33];
  const int k0 = blockIdx.x * 32;
  const int n0 = blockIdx.y * 32;
  const int tid = threadIdx.x;
  {
    const int kr = tid >> 3, nc = (tid & 7) * 4;
    const float4 v = *(const float4*)(in + (size_t)(k0 + kr) * N + n0 + nc);
    ts[kr][nc + 0] = v.x; ts[kr][nc + 1] = v.y;
    ts[kr][nc + 2] = v.z; ts[kr][nc + 3] = v.w;
  }
  __syncthreads();
  {
    const int nr = tid >> 3, kc = (tid & 7) * 4;
    uint2 o;
    o.x = pack_bf16(ts[kc + 0][nr], ts[kc + 1][nr]);
    o.y = pack_bf16(ts[kc + 2][nr], ts[kc + 3][nr]);
    *(uint2*)(outp + (size_t)(n0 + nr) * K + k0 + kc) = o;
  }
}

// Elementwise fp32 -> bf16 (x activations), 4 elements/thread.
__global__ __launch_bounds__(TPB)
void moe_cvt_x(const float* __restrict__ in, unsigned short* __restrict__ outp)
{
  const size_t i = (size_t)blockIdx.x * TPB + threadIdx.x;
  const float4 v = ((const float4*)in)[i];
  uint2 o;
  o.x = pack_bf16(v.x, v.y);
  o.y = pack_bf16(v.z, v.w);
  ((uint2*)outp)[i] = o;
}

// ---------------------------------------------------------------------------
// GEMM1: h[pos,i] = silu(xb[tok]@Wg) * (xb[tok]@Wu) * w[pos]   (bf16 out)
// xb: bf16 [T][H]; wgt/wut: bf16 [I][H] (pre-transposed).
// Async-DMA staging; invalid gather rows clamp to a valid token (finite data)
// and are zeroed by combine-weight 0 in the epilogue.
// ---------------------------------------------------------------------------
__global__ __launch_bounds__(TPB, 1)
void moe_gemm1(const unsigned short* __restrict__ xb_,
               const unsigned short* __restrict__ wgt_,
               const unsigned short* __restrict__ wut_,
               unsigned short* __restrict__ hb_,
               const int* __restrict__ list, const float* __restrict__ wlist,
               const int* __restrict__ cntp, int T, int H, int I)
{
  __shared__ uint lsA[2][BM * PADU];
  __shared__ uint lsBg[2][BN1 * PADU];
  __shared__ uint lsBu[2][BN1 * PADU];
  const __bf16* xb = (const __bf16*)xb_;
  const __bf16* wgt = (const __bf16*)wgt_;
  const __bf16* wut = (const __bf16*)wut_;
  __bf16* hb = (__bf16*)hb_;

  const int cnt = cntp ? cntp[0] : T;
  const int m0 = blockIdx.x * BM;
  if (m0 >= cnt) return;
  const int n0 = blockIdx.y * BN1;
  const int tid = threadIdx.x;
  const int wv = tid >> 5, lane = tid & 31;
  const int half = lane >> 4, lr = lane & 15;

  // A: 128 rows x 4 uint4-chunks (8 bf16 each) = 512 slots; 2 per thread.
  int tokA[2];
#pragma unroll
  for (int it = 0; it < 2; ++it) {
    const int p   = it * TPB + tid;
    const int pos = m0 + (p >> 2);
    const int lp  = (pos < cnt) ? pos : m0;  // clamp to a valid row
    tokA[it] = list ? list[lp] : lp;
  }
  // B: n = tid>>2 (0..63), chunk c = tid&3; same chunk of Wg & Wu.
  const int bn = tid >> 2, bc = tid & 3;

  auto issue_tile = [&](int k0, int buf) {
#pragma unroll
    for (int it = 0; it < 2; ++it) {
      const int p = it * TPB + tid;
      const int row = p >> 2, c = p & 3;
      async_b128(&lsA[buf][row * PADU + c * 4],
                 xb + (size_t)tokA[it] * H + k0 + c * 8);
    }
    const size_t r0 = (size_t)(n0 + bn) * H + k0 + bc * 8;
    async_b128(&lsBg[buf][bn * PADU + bc * 4], wgt + r0);
    async_b128(&lsBu[buf][bn * PADU + bc * 4], wut + r0);
  };

  v8f vz = {};
  v8f accg[4], accu[4];
#pragma unroll
  for (int i = 0; i < 4; ++i) { accg[i] = vz; accu[i] = vz; }

  auto compute = [&](const uint* A, const uint* Bg, const uint* Bu) {
    const v16bf af = frag_a(A, wv * 16 + lr, half);
#pragma unroll
    for (int nt = 0; nt < 4; ++nt) {
      accg[nt] = wmma_bf16(af, frag_b(Bg, nt * 16 + lr, half), accg[nt]);
      accu[nt] = wmma_bf16(af, frag_b(Bu, nt * 16 + lr, half), accu[nt]);
    }
  };

  // Per tile: wait own DMAs -> barrier (publishes tile, fences buf reuse) ->
  // issue next tile's DMAs (other buffer) -> compute.  One barrier per tile.
  issue_tile(0, 0);
  int k0 = 0;
  while (true) {
    wait_async0();
    __syncthreads();
    issue_tile(k0 + BK, 1);                  // always a valid tile
    compute(lsA[0], lsBg[0], lsBu[0]);

    k0 += 2 * BK;
    const bool more = k0 < H;
    wait_async0();
    __syncthreads();
    if (more) issue_tile(k0, 0);
    compute(lsA[1], lsBg[1], lsBu[1]);
    if (!more) break;
  }

  // Epilogue: silu(g)*u*w -> bf16 h. C/D layout: n = lane&15, m = r + half*8.
  float crow[8];
#pragma unroll
  for (int r = 0; r < 8; ++r) {
    const int pos = m0 + wv * 16 + half * 8 + r;
    float w = 1.f;
    if (list) w = (pos < cnt) ? wlist[pos] : 0.f;
    crow[r] = w;
  }
#pragma unroll
  for (int nt = 0; nt < 4; ++nt) {
#pragma unroll
    for (int r = 0; r < 8; ++r) {
      const float g = accg[nt][r], u = accu[nt][r];
      const float s = g / (1.f + __expf(-g));
      const float hv = s * u * crow[r];
      const int pos = m0 + wv * 16 + half * 8 + r;
      const int col = n0 + nt * 16 + lr;
      hb[(size_t)pos * I + col] = (__bf16)hv;
    }
  }
}

// ---------------------------------------------------------------------------
// GEMM2: out[tok,:] (+)= h[pos,:] @ Wdown    (scatter by gather list)
// hb: bf16 [T][I]; wdt: bf16 [H][I] (pre-transposed). Async-DMA staging.
// ---------------------------------------------------------------------------
__global__ __launch_bounds__(TPB, 1)
void moe_gemm2(const unsigned short* __restrict__ hb_,
               const unsigned short* __restrict__ wdt_,
               float* __restrict__ out, const int* __restrict__ list,
               const int* __restrict__ cntp, int T, int I, int H, int accum)
{
  __shared__ uint lsA[2][BM * PADU];
  __shared__ uint lsB[2][BN2 * PADU];
  const __bf16* hb = (const __bf16*)hb_;
  const __bf16* wdt = (const __bf16*)wdt_;

  const int cnt = cntp ? cntp[0] : T;
  const int m0 = blockIdx.x * BM;
  if (m0 >= cnt) return;
  const int n0 = blockIdx.y * BN2;
  const int tid = threadIdx.x;
  const int wv = tid >> 5, lane = tid & 31;
  const int half = lane >> 4, lr = lane & 15;

  auto issue_tile = [&](int k0, int buf) {
#pragma unroll
    for (int it = 0; it < 2; ++it) {
      const int p = it * TPB + tid;
      const int row = p >> 2, c = p & 3;
      async_b128(&lsA[buf][row * PADU + c * 4],
                 hb + (size_t)(m0 + row) * I + k0 + c * 8);
      async_b128(&lsB[buf][row * PADU + c * 4],
                 wdt + (size_t)(n0 + row) * I + k0 + c * 8);
    }
  };

  v8f vz = {};
  v8f acc[8];
#pragma unroll
  for (int i = 0; i < 8; ++i) acc[i] = vz;

  auto compute = [&](const uint* A, const uint* B) {
    const v16bf af = frag_a(A, wv * 16 + lr, half);
#pragma unroll
    for (int nt = 0; nt < 8; ++nt)
      acc[nt] = wmma_bf16(af, frag_b(B, nt * 16 + lr, half), acc[nt]);
  };

  issue_tile(0, 0);
  int k0 = 0;
  while (true) {
    wait_async0();
    __syncthreads();
    issue_tile(k0 + BK, 1);                  // always a valid tile
    compute(lsA[0], lsB[0]);

    k0 += 2 * BK;
    const bool more = k0 < I;
    wait_async0();
    __syncthreads();
    if (more) issue_tile(k0, 0);
    compute(lsA[1], lsB[1]);
    if (!more) break;
  }

#pragma unroll
  for (int nt = 0; nt < 8; ++nt) {
#pragma unroll
    for (int r = 0; r < 8; ++r) {
      const int pos = m0 + wv * 16 + half * 8 + r;
      if (pos < cnt) {
        const int tok = list ? list[pos] : pos;
        const int col = n0 + nt * 16 + lr;
        const size_t o = (size_t)tok * H + col;
        const float v = acc[nt][r];
        out[o] = accum ? (out[o] + v) : v;   // passes are stream-serialized
      }
    }
  }
}

// ---------------------------------------------------------------------------
extern "C" void kernel_launch(void* const* d_in, const int* in_sizes, int n_in,
                              void* d_out, int out_size, void* d_ws, size_t ws_size,
                              hipStream_t stream) {
  const float* x      = (const float*)d_in[0];
  const float* gate_w = (const float*)d_in[1];
  const float* ebias  = (const float*)d_in[2];
  const float* swg    = (const float*)d_in[3];
  const float* swu    = (const float*)d_in[4];
  const float* swd    = (const float*)d_in[5];
  const float* rwg    = (const float*)d_in[6];
  const float* rwu    = (const float*)d_in[7];
  const float* rwd    = (const float*)d_in[8];

  const int E = in_sizes[2];            // 8
  const int H = in_sizes[1] / E;        // 1024
  const int T = in_sizes[0] / H;        // 2048
  const int I = in_sizes[3] / H;        // 1024

  float* out    = (float*)d_out;
  int*  idx_out = (int*)((char*)d_out + (size_t)T * H * sizeof(float));

  char* ws = (char*)d_ws;
  size_t off = 0;
  int* counts = (int*)(ws + off);                   off += 256;
  int* list   = (int*)(ws + off);                   off += (size_t)E * T * 4;
  float* wlist = (float*)(ws + off);                off += (size_t)E * T * 4;
  unsigned short* xb  = (unsigned short*)(ws + off); off += (size_t)T * H * 2;
  unsigned short* wtg = (unsigned short*)(ws + off); off += (size_t)I * H * 2;
  unsigned short* wtu = (unsigned short*)(ws + off); off += (size_t)I * H * 2;
  unsigned short* wtd = (unsigned short*)(ws + off); off += (size_t)H * I * 2;
  unsigned short* hb  = (unsigned short*)(ws + off); off += (size_t)T * I * 2;

  moe_zero_counts<<<1, 32, 0, stream>>>(counts);
  moe_router<<<T / (TPB / 32), TPB, 0, stream>>>(x, gate_w, ebias, counts, list,
                                                 wlist, idx_out, T, H);
  moe_cvt_x<<<(int)(((size_t)T * H / 4) / TPB), TPB, 0, stream>>>(x, xb);

  dim3 gtGU(H / 32, I / 32);  // Wg/Wu: [H][I] -> [I][H]
  dim3 gtD (I / 32, H / 32);  // Wd:    [I][H] -> [H][I]
  dim3 g1(T / BM, I / BN1);   // 16 x 16
  dim3 g2(T / BM, H / BN2);   // 16 x 8

  // Shared expert: full write (no zero-init of out needed)
  moe_cvt_tr<<<gtGU, TPB, 0, stream>>>(swg, wtg, H, I);
  moe_cvt_tr<<<gtGU, TPB, 0, stream>>>(swu, wtu, H, I);
  moe_cvt_tr<<<gtD,  TPB, 0, stream>>>(swd, wtd, I, H);
  moe_gemm1<<<g1, TPB, 0, stream>>>(xb, wtg, wtu, hb, nullptr, nullptr, nullptr,
                                    T, H, I);
  moe_gemm2<<<g2, TPB, 0, stream>>>(hb, wtd, out, nullptr, nullptr, T, I, H, 0);

  // Routed experts: gathered top-2 dispatch, accumulate
  for (int e = 0; e < E; ++e) {
    moe_cvt_tr<<<gtGU, TPB, 0, stream>>>(rwg + (size_t)e * H * I, wtg, H, I);
    moe_cvt_tr<<<gtGU, TPB, 0, stream>>>(rwu + (size_t)e * H * I, wtu, H, I);
    moe_cvt_tr<<<gtD,  TPB, 0, stream>>>(rwd + (size_t)e * I * H, wtd, I, H);
    moe_gemm1<<<g1, TPB, 0, stream>>>(xb, wtg, wtu, hb,
                                      list + (size_t)e * T,
                                      wlist + (size_t)e * T, counts + e,
                                      T, H, I);
    moe_gemm2<<<g2, TPB, 0, stream>>>(hb, wtd, out,
                                      list + (size_t)e * T, counts + e,
                                      T, I, H, 1);
  }
}
